// Sparse3DNA_22668837388652
// MI455X (gfx1250) — matrix operand order repaired and checked by reading, not compiled, not run on hardware
//
#include <hip/hip_runtime.h>
#include <hip/hip_bf16.h>
#include <cstdint>

// ---------------------------------------------------------------------------
// Sparse3DNA on MI455X (gfx1250, wave32)
//   DIM=512, HEADS=8, DIM_HEAD=64, VID=(8,32,32), KS=3^3 -> 28 keys/token
//   n = 8193 rows (BOS + 8192 tokens)
// GEMMs: v_wmma_f32_16x16x32_bf16 with A slab staged in LDS by the Tensor
// Data Mover (tensor_load_to_lds + s_wait_tensorcnt), each wave computing a
// 16x64 strip (one A fragment amortized over 4 wmma per K-step).
// Attention: fp32 VALU, L2-resident K/V gathers, talking-heads via wave32
// shuffles.
// ---------------------------------------------------------------------------

typedef __attribute__((ext_vector_type(16))) __bf16 v16bf;
typedef __attribute__((ext_vector_type(8)))  __bf16 v8bf;
typedef __attribute__((ext_vector_type(8)))  float  v8f;
typedef unsigned int u32x4 __attribute__((ext_vector_type(4)));
typedef int          i32x8 __attribute__((ext_vector_type(8)));
typedef int          i32x4 __attribute__((ext_vector_type(4)));

#define NTOK   8192
#define NROWS  8193      // BOS + tokens
#define DIMM   512
#define NQKV   1536      // q|k|v columns
#define DHEAD  64
#define KDIM   512       // K of both GEMMs
#define LDSPAD 8         // bf16 elems (16B) pad per LDS row -> conflict-free ds_load
#define LDA    (KDIM + LDSPAD)

// ---------------------------------------------------------------- converts --
__global__ void convert_x_kernel(const float* __restrict__ x,
                                 __hip_bfloat16* __restrict__ xb, int n) {
  int i = blockIdx.x * blockDim.x + threadIdx.x;
  if (i < n) xb[i] = __float2bfloat16(x[i]);
}

// Wqkv_t[c][k] = c<512 ? w_q[k][c] * d^-0.5 : w_kv[k][c-512]   (N-major, K-contig)
__global__ void pack_wqkv_kernel(const float* __restrict__ w_q,
                                 const float* __restrict__ w_kv,
                                 __hip_bfloat16* __restrict__ wt) {
  int i = blockIdx.x * blockDim.x + threadIdx.x;
  if (i >= NQKV * DIMM) return;
  int c = i / DIMM, k = i - c * DIMM;
  float v = (c < DIMM) ? w_q[(size_t)k * DIMM + c] * 0.125f   // 64^-0.5
                       : w_kv[(size_t)k * (2 * DIMM) + (c - DIMM)];
  wt[(size_t)c * DIMM + k] = __float2bfloat16(v);
}

__global__ void pack_wout_kernel(const float* __restrict__ w_out,
                                 __hip_bfloat16* __restrict__ wt) {
  int i = blockIdx.x * blockDim.x + threadIdx.x;
  if (i >= DIMM * DIMM) return;
  int c = i / DIMM, k = i - c * DIMM;
  wt[(size_t)c * DIMM + k] = __float2bfloat16(w_out[(size_t)k * DIMM + c]);
}

// ------------------------------------------------------------- WMMA GEMM ----
// C[M,N] = A[M,K](bf16, row-major) * Bt[N,K](bf16, pre-transposed) (+ bias)
// Block: 256 thr = 8 waves, 16 rows x 512 cols of C; wave owns 16x64 strip.
// A slab (16 x 512 bf16) staged into LDS by TDM (one descriptor per block),
// rows are 16B-padded so the 16 parallel row reads hit distinct banks.
// Fragment layouts per CDNA5 ISA 7.12.2 (wave32):
//   A 16x32 : row = lane&15; elems 0-7 => K=(lane>>4)*8+e, 8-15 => +16
//   B 32x16 : col = lane&15; elems 0-7 => K=(lane>>4)*8+e, 8-15 => +16
//   C 16x16 : col = lane&15; acc[r]  => row r + 8*(lane>>4)
__global__ void gemm_bf16_wmma_kernel(const __hip_bfloat16* __restrict__ A,
                                      const __hip_bfloat16* __restrict__ Bt,
                                      float* __restrict__ C,
                                      const float* __restrict__ bias,
                                      int M, int N) {
  __shared__ __align__(16) __bf16 sA[16 * LDA];

  const int lane = threadIdx.x & 31;
  const int wave = threadIdx.x >> 5;
  const int half = lane >> 4;
  const int sub  = lane & 15;
  const int mbase = blockIdx.y * 16;

#if __has_builtin(__builtin_amdgcn_tensor_load_to_lds)
  if (wave == 0) {
    // Tensor DMA descriptor (CDNA5 ISA ch.8): 2D bf16 tensor, tile 512 x 16,
    // tensor_dim1 = rows remaining so the ragged last tile zero-fills,
    // LDS pad: 4 dwords after every 256 dwords (one 512-elem K row).
    const unsigned long long gaddr =
        (unsigned long long)(uintptr_t)(A + (size_t)mbase * KDIM);
    const unsigned rows_rem = (unsigned)(M - mbase);
    u32x4 g0;
    g0[0] = 1u;                                         // count = 1 descriptor
    g0[1] = (unsigned)(uintptr_t)sA;                    // lds_addr (bytes)
    g0[2] = (unsigned)(gaddr & 0xFFFFFFFFull);          // global_addr[31:0]
    g0[3] = (unsigned)((gaddr >> 32) & 0x1FFFFFFull)    // global_addr[56:32]
          | (2u << 30);                                 // type = 2 (image)
    i32x8 g1;
    g1[0] = (int)((1u << 16)       // data_size: 2 bytes
                | (1u << 20)       // pad_enable
                | (7u << 22)       // pad_interval: 256 dwords
                | (3u << 25));     // pad_amount: 4 dwords (16B)
    g1[1] = (int)(((unsigned)KDIM & 0xFFFFu) << 16);                 // tdim0 lo
    g1[2] = (int)(((unsigned)KDIM >> 16) | ((rows_rem & 0xFFFFu) << 16));
    g1[3] = (int)((rows_rem >> 16) | ((unsigned)KDIM << 16));        // tile_dim0
    g1[4] = 16;                                         // tile_dim1 (tile_dim2=0)
    g1[5] = KDIM;                                       // tensor_dim0_stride lo
    g1[6] = 0;
    g1[7] = 0;
    i32x4 z4 = {};
#if __clang_major__ >= 23
    i32x8 z8 = {};
    __builtin_amdgcn_tensor_load_to_lds(g0, g1, z4, z4, z8, 0);
#else
    __builtin_amdgcn_tensor_load_to_lds(g0, g1, z4, z4, 0);
#endif
    __builtin_amdgcn_s_wait_tensorcnt((short)0);
  }
#else
  // Fallback: cooperative vectorized copy (same padded LDS layout).
  for (int idx = threadIdx.x; idx < 16 * (KDIM / 8); idx += 256) {
    int r = idx / (KDIM / 8), c8 = idx - r * (KDIM / 8);
    int row = mbase + r; if (row >= M) row = M - 1;
    *(v8bf*)&sA[r * LDA + c8 * 8] =
        *(const v8bf*)((const __bf16*)A + (size_t)row * KDIM + c8 * 8);
  }
#endif
  __syncthreads();

  const int ncol0 = (blockIdx.x * 8 + wave) * 64;          // 16x64 strip
  const __bf16* lA = sA + sub * LDA + half * 8;
  const __bf16* Bb = (const __bf16*)Bt + (size_t)(ncol0 + sub) * KDIM + half * 8;

  v8f acc[4] = {};
  for (int kb = 0; kb < KDIM; kb += 32) {
    __builtin_prefetch(Bb + kb + 32, 0, 3);                // global_prefetch_b8
    v8bf alo = *(const v8bf*)(lA + kb);                    // ds_load_b128
    v8bf ahi = *(const v8bf*)(lA + kb + 16);
    v16bf af;
#pragma unroll
    for (int e = 0; e < 8; ++e) { af[e] = alo[e]; af[8 + e] = ahi[e]; }
#pragma unroll
    for (int t = 0; t < 4; ++t) {
      const __bf16* bp = Bb + (size_t)t * 16 * KDIM + kb;
      v8bf blo = *(const v8bf*)bp;
      v8bf bhi = *(const v8bf*)(bp + 16);
      v16bf bfr;
#pragma unroll
      for (int e = 0; e < 8; ++e) { bfr[e] = blo[e]; bfr[8 + e] = bhi[e]; }
      acc[t] = __builtin_amdgcn_wmma_f32_16x16x32_bf16(
          false, af, false, bfr, (short)0, acc[t], false, false);
    }
  }

#pragma unroll
  for (int t = 0; t < 4; ++t) {
    const int col = ncol0 + t * 16 + sub;
    const float b = bias ? bias[col] : 0.f;
#pragma unroll
    for (int r = 0; r < 8; ++r) {
      int row = mbase + half * 8 + r;
      if (row < M) C[(size_t)row * N + col] = acc[t][r] + b;
    }
  }
}

// ------------------------------------------------------------- attention ----
// qkv[8193][1536] fp32: cols [0,512)=q(prescaled), [512,1024)=k, [1024,1536)=v
// lane = (token-slot, head): 8 consecutive lanes = 8 heads of one token.
__global__ void attn_kernel(const float* __restrict__ qkv,
                            const float* __restrict__ w_th,
                            __hip_bfloat16* __restrict__ oi) {
  __shared__ float sW[64];
  if (threadIdx.x < 64) sW[threadIdx.x] = w_th[threadIdx.x];
  __syncthreads();

  // BOS output row = v_bos
  if (blockIdx.x == 0)
    for (int c = threadIdx.x; c < DIMM; c += 256)
      oi[c] = __float2bfloat16(qkv[2 * DIMM + c]);

  const int h = threadIdx.x & 7;
  const int i = blockIdx.x * 32 + (threadIdx.x >> 3);   // token 0..8191
  const int f = i >> 10, y = (i >> 5) & 31, xc = i & 31;

  float wrow[8];
#pragma unroll
  for (int t = 0; t < 8; ++t) wrow[t] = sW[h * 8 + t];  // w_th[g=h][:]

  // neighbor rows in qkv (+1 for BOS offset); -1 = masked
  int tok[28];
  tok[0] = 0;                                           // BOS key, never masked
#pragma unroll
  for (int j = 1; j < 28; ++j) {
    int a = (j - 1) / 9, bq = ((j - 1) / 3) % 3, cq = (j - 1) % 3;
    int sf = f + a - 1, sy = y + bq - 1, sx = xc + cq - 1;
    int t = (sf * 32 + sy) * 32 + sx;
    bool ok = (sf >= 0) & (sf < 8) & (sy >= 0) & (sy < 32) &
              (sx >= 0) & (sx < 32) && (t <= i);        // causal
    tok[j] = ok ? (t + 1) : -1;
  }

  float q[64];
  const float4* qp = (const float4*)(qkv + (size_t)(i + 1) * NQKV + h * DHEAD);
#pragma unroll
  for (int dq = 0; dq < 16; ++dq) {
    float4 v = qp[dq];
    q[4*dq] = v.x; q[4*dq+1] = v.y; q[4*dq+2] = v.z; q[4*dq+3] = v.w;
  }

  float s[28];
  float mx = -3.0e38f;
#pragma unroll
  for (int j = 0; j < 28; ++j) {
    float dot = -3.0e38f;
    if (tok[j] >= 0) {
      dot = 0.f;
      const float4* kp = (const float4*)(qkv + (size_t)tok[j] * NQKV + DIMM + h * DHEAD);
#pragma unroll
      for (int dq = 0; dq < 16; ++dq) {
        float4 kv = kp[dq];
        dot += q[4*dq]*kv.x + q[4*dq+1]*kv.y + q[4*dq+2]*kv.z + q[4*dq+3]*kv.w;
      }
      mx = fmaxf(mx, dot);
    }
    s[j] = dot;
  }
  float denom = 0.f;
#pragma unroll
  for (int j = 0; j < 28; ++j) {
    float p = (tok[j] >= 0) ? __expf(s[j] - mx) : 0.f;
    s[j] = p; denom += p;
  }
  const float inv = 1.f / denom;
#pragma unroll
  for (int j = 0; j < 28; ++j) s[j] *= inv;

  // talking heads: mix softmaxed attn across the 8 head-lanes of this token
  const int lbase = (threadIdx.x & 31) & ~7;
  float ns[28];
#pragma unroll
  for (int j = 0; j < 28; ++j) {
    float aj = s[j], m = 0.f;
#pragma unroll
    for (int hh = 0; hh < 8; ++hh)
      m += wrow[hh] * __shfl(aj, lbase + hh, 32);
    ns[j] = m;
  }

  float o[64];
#pragma unroll
  for (int d = 0; d < 64; ++d) o[d] = 0.f;
#pragma unroll
  for (int j = 0; j < 28; ++j) {
    if (tok[j] < 0) continue;                 // masked in every head -> weight 0
    const float w = ns[j];
    const float4* vp = (const float4*)(qkv + (size_t)tok[j] * NQKV + 2 * DIMM + h * DHEAD);
#pragma unroll
    for (int dq = 0; dq < 16; ++dq) {
      float4 vv = vp[dq];
      o[4*dq] += w*vv.x; o[4*dq+1] += w*vv.y; o[4*dq+2] += w*vv.z; o[4*dq+3] += w*vv.w;
    }
  }
  __hip_bfloat16* op = oi + (size_t)(i + 1) * DIMM + h * DHEAD;
#pragma unroll
  for (int d = 0; d < 64; ++d) op[d] = __float2bfloat16(o[d]);
}

// ---------------------------------------------------------------- launch ----
extern "C" void kernel_launch(void* const* d_in, const int* in_sizes, int n_in,
                              void* d_out, int out_size, void* d_ws, size_t ws_size,
                              hipStream_t stream) {
  const float* x     = (const float*)d_in[0];
  const float* w_q   = (const float*)d_in[1];
  const float* w_kv  = (const float*)d_in[2];
  const float* w_th  = (const float*)d_in[3];
  const float* w_out = (const float*)d_in[4];
  const float* b_out = (const float*)d_in[5];
  float* out = (float*)d_out;

  // workspace layout (256B-aligned slabs, ~66 MB total)
  char* ws = (char*)d_ws;
  size_t off = 0;
  auto take = [&](size_t bytes) { char* p = ws + off; off += (bytes + 255) & ~(size_t)255; return p; };
  __hip_bfloat16* xb    = (__hip_bfloat16*)take((size_t)NROWS * DIMM * 2);
  __hip_bfloat16* wqkvt = (__hip_bfloat16*)take((size_t)NQKV * DIMM * 2);
  __hip_bfloat16* woutt = (__hip_bfloat16*)take((size_t)DIMM * DIMM * 2);
  float*          qkv   = (float*)take((size_t)NROWS * NQKV * 4);
  __hip_bfloat16* oi    = (__hip_bfloat16*)take((size_t)NROWS * DIMM * 2);
  (void)ws_size; (void)in_sizes; (void)n_in; (void)out_size;

  // 1) precision conversion + weight packing (transposed for WMMA B frags)
  {
    int n = NROWS * DIMM;
    convert_x_kernel<<<(n + 255) / 256, 256, 0, stream>>>(x, xb, n);
    pack_wqkv_kernel<<<(NQKV * DIMM + 255) / 256, 256, 0, stream>>>(w_q, w_kv, wqkvt);
    pack_wout_kernel<<<(DIMM * DIMM + 255) / 256, 256, 0, stream>>>(w_out, woutt);
  }

  // 2) fused QKV projection: qkv = xb @ [wq*scale | wkv]   (bf16 WMMA, f32 acc)
  gemm_bf16_wmma_kernel<<<dim3(NQKV / 512, (NROWS + 15) / 16), 256, 0, stream>>>(
      xb, wqkvt, qkv, nullptr, NROWS, NQKV);

  // 3) sparse 3D neighborhood attention + talking heads -> bf16 inner output
  attn_kernel<<<NTOK / 32, 256, 0, stream>>>(qkv, w_th, oi);

  // 4) output projection with bias -> fp32 result
  gemm_bf16_wmma_kernel<<<dim3(DIMM / 512, (NROWS + 15) / 16), 256, 0, stream>>>(
      oi, woutt, out, b_out, NROWS, DIMM);
}
